// LinearTernary_10436770529924
// MI455X (gfx1250) — matrix operand order repaired and checked by reading, not compiled, not run on hardware
//
#include <hip/hip_runtime.h>

// ---------------------------------------------------------------------------
// Ternary (2-bit) quantized GEMM for MI455X (gfx1250, wave32, WMMA).
//   out[M,N] = fp16( fp32( x[M,K] @ dequant(Wp, S)[K,N] ) ) + bias[N]
// Compute-bound (739 GFLOP vs ~255 MB HBM floor; x/weights stay L2-resident).
// Block: 512 threads (16 wave32s), tile 256x128, BK=32.
//  - A tile staged HBM -> LDS via gfx1250 async-to-LDS (no VGPR staging, no
//    spills), synced with s_wait_asynccnt + barrier.
//  - Weights dequantized in-register to fp16 bit patterns {s^0x8000, 0, s},
//    8 codes per thread (balanced across all 16 waves), stored transposed
//    [n][k] so each thread writes one contiguous b128.
// ---------------------------------------------------------------------------

typedef _Float16 half_t;
typedef __attribute__((ext_vector_type(16))) _Float16 v16h;
typedef __attribute__((ext_vector_type(8)))  float    v8f;
typedef __attribute__((ext_vector_type(4)))  int      v4i;

#define K_DIM 4096
#define N_DIM 11008
#define M_DIM 8192
#define GROUP 128

#define BM 256
#define BN 128
#define BK 32                  // one WMMA k=32 step per iteration
#define NTHREADS 512           // 16 wave32s
#define KSTEPS (K_DIM / BK)    // 128

#if __has_builtin(__builtin_amdgcn_global_load_async_to_lds_b128)
#define USE_ASYNC_LDS 1
#else
#define USE_ASYNC_LDS 0
#endif

// exact pointer types for the async builtin: (v4i* src-global, v4i* dst-lds)
typedef __attribute__((address_space(1))) v4i* gv4_t;
typedef __attribute__((address_space(3))) v4i* lv4_t;

__global__ __launch_bounds__(NTHREADS)
void ternary_wmma_gemm(const half_t* __restrict__ X,          // [M,K] fp16
                       const int*    __restrict__ Wp,         // [K/16, N] int32 codes
                       const unsigned short* __restrict__ S,  // [K/128, N] fp16 bits
                       const half_t* __restrict__ Bias,       // [N] fp16
                       half_t*       __restrict__ Out)        // [M,N] fp16
{
    // Double-buffered LDS. Bt transposed [n][k]: each thread's 8 dequantized
    // codes form one contiguous b128; each WMMA B-fragment lane reads its
    // column's k {8h..8h+7, 16+8h..23+8h} as two b128 reads.
    __shared__ half_t As[2][BM][BK];   // 2 x 16 KB
    __shared__ half_t Bt[2][BN][BK];   // 2 x  8 KB   (48 KB total)

    const int tid  = threadIdx.x;
    const int lane = tid & 31;
    const int wave = tid >> 5;      // 16 waves
    const int wm   = wave >> 2;     // 0..3 -> 64 rows each
    const int wn   = wave & 3;      // 0..3 -> 32 cols each
    const int m0   = blockIdx.y * BM;
    const int n0   = blockIdx.x * BN;

    const int hi = lane >> 4;       // half-wave select per ISA VGPR layouts
    const int lo = lane & 15;

    v8f acc[4][2];
#pragma unroll
    for (int mt = 0; mt < 4; ++mt)
#pragma unroll
        for (int nt = 0; nt < 2; ++nt)
            acc[mt][nt] = (v8f){0.f, 0.f, 0.f, 0.f, 0.f, 0.f, 0.f, 0.f};

    // ---- A tile: 256 rows x 32 halfs = 1024 uint4, 2 async loads/thread ----
    auto issueA = [&](int kk, int buf) {
#pragma unroll
        for (int i = 0; i < 2; ++i) {
            int idx = tid + i * NTHREADS;
            int row = idx >> 2;          // 4 uint4 (32 halfs) per row
            int c   = idx & 3;
            const half_t* src = X + (size_t)(m0 + row) * K_DIM + kk * BK + c * 8;
#if USE_ASYNC_LDS
            __builtin_amdgcn_global_load_async_to_lds_b128(
                (gv4_t)src, (lv4_t)&As[buf][row][c * 8], 0, 0);
#else
            *(uint4*)(&As[buf][row][c * 8]) = *(const uint4*)src;
#endif
        }
    };

    // ---- W tile: 2 packed rows x 128 cols = 256 int32; every thread
    //      dequantizes 8 codes (half an int32) -> one b128 store ----
    const int wi    = tid >> 1;      // int32 index 0..255
    const int wpr   = wi >> 7;       // packed row 0..1 within K-step
    const int wcol  = wi & 127;      // column within N tile
    const int whalf = tid & 1;       // which 8 codes of the dword

    int            wReg = 0;
    unsigned short sReg = 0;
    auto loadW = [&](int kk) {
        wReg = Wp[(size_t)(kk * 2 + wpr) * N_DIM + n0 + wcol];  // pair-coalesced
        sReg = S[(size_t)((kk * BK) / GROUP) * N_DIM + n0 + wcol];
    };
    auto commitW = [&](int buf) {
        const unsigned w = ((unsigned)wReg) >> (16 * whalf);    // my 8 codes
        const unsigned short sp = sReg;            // +s (fp16 bits)
        const unsigned short sn = sReg ^ 0x8000u;  // -s (fp16 bits)
        union { uint4 u; unsigned short h[8]; } d;
#pragma unroll
        for (int j = 0; j < 8; ++j) {              // 0 -> -s, 1 -> 0, 2 -> +s
            unsigned c2 = (w >> (2 * j)) & 3u;
            d.h[j] = (c2 & 2u) ? sp : ((c2 & 1u) ? (unsigned short)0 : sn);
        }
        *(uint4*)(&Bt[buf][wcol][wpr * 16 + whalf * 8]) = d.u;
    };

    auto computeTile = [&](int buf) {
        v16h afrag[4], bfrag[2];
        // A fragment (16x32 f16): lane lo = row; lanes<16 hold K {0..7,16..23},
        // lanes>=16 hold K {8..15,24..31} -> two b128 reads per fragment.
#pragma unroll
        for (int mt = 0; mt < 4; ++mt) {
            union { uint4 u[2]; v16h v; } t;
            const half_t* p = &As[buf][wm * 64 + mt * 16 + lo][hi * 8];
            t.u[0] = *(const uint4*)p;
            t.u[1] = *(const uint4*)(p + 16);
            afrag[mt] = t.v;
        }
        // B fragment (32x16 f16): lane lo = column; lanes<16 hold K {0..7,16..23},
        // lanes>=16 hold K {8..15,24..31} -> two b128 reads from transposed tile.
#pragma unroll
        for (int nt = 0; nt < 2; ++nt) {
            union { uint4 u[2]; v16h v; } t;
            const half_t* p = &Bt[buf][wn * 32 + nt * 16 + lo][hi * 8];
            t.u[0] = *(const uint4*)p;
            t.u[1] = *(const uint4*)(p + 16);
            bfrag[nt] = t.v;
        }
#pragma unroll
        for (int mt = 0; mt < 4; ++mt)
#pragma unroll
            for (int nt = 0; nt < 2; ++nt)
                acc[mt][nt] = __builtin_amdgcn_wmma_f32_16x16x32_f16(
                    false, afrag[mt], false, bfrag[nt],
                    (short)0, acc[mt][nt], false, false);
    };

    auto waitAsync = [&]() {
#if USE_ASYNC_LDS
#if __has_builtin(__builtin_amdgcn_s_wait_asynccnt)
        __builtin_amdgcn_s_wait_asynccnt(0);
#else
        asm volatile("s_wait_asynccnt 0x0" ::: "memory");
#endif
#endif
    };

    // ---- prologue: tile 0 into buffer 0 ----
    issueA(0, 0);
    loadW(0);
    commitW(0);
    waitAsync();
    __syncthreads();

    // ---- pipelined main loop: stage k+1 while computing k ----
    for (int kk = 0; kk < KSTEPS; ++kk) {
        const int cur = kk & 1;
        const int nxt = cur ^ 1;
        if (kk + 1 < KSTEPS) {
            issueA(kk + 1, nxt);   // async: HBM -> LDS, no VGPRs
            loadW(kk + 1);
        }
        computeTile(cur);
        if (kk + 1 < KSTEPS) commitW(nxt);
        waitAsync();
        __syncthreads();
    }

    // ---- epilogue: C/D layout -> lane lo = N, VGPR r = M row (r + 8*hi) ----
    // Output (180 MB) is pure streaming: non-temporal stores keep the
    // L2-resident x / packed weights from being evicted.
#pragma unroll
    for (int nt = 0; nt < 2; ++nt) {
        const int n = n0 + wn * 32 + nt * 16 + lo;
        const half_t b = Bias[n];
#pragma unroll
        for (int mt = 0; mt < 4; ++mt) {
            const int mbase = m0 + wm * 64 + mt * 16 + hi * 8;
#pragma unroll
            for (int r = 0; r < 8; ++r) {
                half_t v = (half_t)acc[mt][nt][r] + b;   // fp32->fp16, then +bias
                __builtin_nontemporal_store(v, &Out[(size_t)(mbase + r) * N_DIM + n]);
            }
        }
    }
}

extern "C" void kernel_launch(void* const* d_in, const int* in_sizes, int n_in,
                              void* d_out, int out_size, void* d_ws, size_t ws_size,
                              hipStream_t stream) {
    const half_t*         x    = (const half_t*)d_in[0];          // fp16 [2,4096,4096]
    const int*            wp   = (const int*)d_in[1];             // int32 [256, 11008]
    const unsigned short* s    = (const unsigned short*)d_in[2];  // fp16 bits [32, 11008]
    const half_t*         bias = (const half_t*)d_in[3];          // fp16 [11008]
    half_t*               out  = (half_t*)d_out;                  // fp16 [2,4096,11008]

    dim3 grid(N_DIM / BN, M_DIM / BM);  // 86 x 32
    dim3 block(NTHREADS);
    ternary_wmma_gemm<<<grid, block, 0, stream>>>(x, wp, s, bias, out);
}
